// GNNLayer_73770358276178
// MI455X (gfx1250) — matrix operand order repaired and checked by reading, not compiled, not run on hardware
//
#include <hip/hip_runtime.h>
#include <hip/hip_bf16.h>

typedef __bf16 bf16_t;
typedef __attribute__((ext_vector_type(16))) __bf16 v16bf;
typedef __attribute__((ext_vector_type(8)))  float  v8f;

constexpr int B_ = 16, NV_ = 8192, NC_ = 4096, D_ = 128, E_ = 49152;

// ---------------------------------------------------------------------------
// Branch-free fast transcendentals (no EXEC divergence, no IEEE div sequences:
// v_rcp_f32 instead of v_div_scale/v_div_fmas chains).
// ---------------------------------------------------------------------------
__device__ __forceinline__ float fast_rcp(float x) {
  return __builtin_amdgcn_rcpf(x);
}
__device__ __forceinline__ float fast_sigmoid(float x) {
  return fast_rcp(1.0f + __expf(-x));
}
__device__ __forceinline__ float fast_tanh(float x) {
  // 1 - 2/(1+exp(2x)); saturates correctly as exp overflows/underflows.
  return fmaf(-2.0f, fast_rcp(1.0f + __expf(2.0f * x)), 1.0f);
}
// Abramowitz-Stegun 7.1.26 erf, |err| <= 1.5e-7, branch-free.
__device__ __forceinline__ float fast_erf(float x) {
  const float ax = fabsf(x);
  const float t = fast_rcp(fmaf(0.3275911f, ax, 1.0f));
  float p = fmaf(1.061405429f, t, -1.453152027f);
  p = fmaf(p, t, 1.421413741f);
  p = fmaf(p, t, -0.284496736f);
  p = fmaf(p, t, 0.254829592f);
  const float y = 1.0f - p * t * __expf(-ax * ax);
  return copysignf(y, x);
}
__device__ __forceinline__ float fast_gelu(float x) {
  return 0.5f * x * (1.0f + fast_erf(x * 0.7071067811865475f));
}

// ---------------------------------------------------------------------------
// Build one A-operand chunk (16x32, bf16) for v_wmma_f32_16x16x32_bf16.
// Layout: lanes 0-15 = row M, elements 0..7 -> K {0..7}, 8..15 -> K {16..23};
// lanes 16-31: K {8..15} and {24..31}.  p points at X[row*ld + chunk*32].
// o = (lane>=16) ? 8 : 0.
// ---------------------------------------------------------------------------
__device__ __forceinline__ v16bf load_a16(const float* p, int o) {
  float4 fa = *(const float4*)(p + o);
  float4 fb = *(const float4*)(p + o + 4);
  float4 fc = *(const float4*)(p + o + 16);
  float4 fd = *(const float4*)(p + o + 20);
  v16bf v;
  v[0]  = (__bf16)fa.x; v[1]  = (__bf16)fa.y; v[2]  = (__bf16)fa.z; v[3]  = (__bf16)fa.w;
  v[4]  = (__bf16)fb.x; v[5]  = (__bf16)fb.y; v[6]  = (__bf16)fb.z; v[7]  = (__bf16)fb.w;
  v[8]  = (__bf16)fc.x; v[9]  = (__bf16)fc.y; v[10] = (__bf16)fc.z; v[11] = (__bf16)fc.w;
  v[12] = (__bf16)fd.x; v[13] = (__bf16)fd.y; v[14] = (__bf16)fd.z; v[15] = (__bf16)fd.w;
  return v;
}

// One 16x16 output tile = sum over 4 K-chunks (K=128).
__device__ __forceinline__ v8f gemm4(const v16bf* A, const bf16_t* __restrict__ pk,
                                     int j, int lane) {
  v8f acc = {0.f, 0.f, 0.f, 0.f, 0.f, 0.f, 0.f, 0.f};
#pragma unroll
  for (int c = 0; c < 4; ++c) {
    const v16bf bv = *(const v16bf*)(pk + ((((j * 4 + c) * 32) + lane) << 4));
    acc = __builtin_amdgcn_wmma_f32_16x16x32_bf16(
        /*neg_a=*/false, A[c], /*neg_b=*/false, bv,
        /*c_mod=*/(short)0, acc, /*reuse_a=*/false, /*reuse_b=*/false);
  }
  return acc;
}

// ---------------------------------------------------------------------------
// Pack W (N x 128, row-major, computing x @ W^T) into WMMA B-operand order.
// B layout (16-bit, 32x16): lanes 0-15 = column N, i 0..7 -> K {0..7}, 8..15 -> K {16..23};
// lanes 16-31: K {8..15} and {24..31} (plus chunk*32).
// pk flat index = ((j*4 + c)*32 + lane)*16 + i.
// ---------------------------------------------------------------------------
__global__ void pack_w(const float* __restrict__ W, bf16_t* __restrict__ pk, int total) {
  int idx = blockIdx.x * blockDim.x + threadIdx.x;
  if (idx >= total) return;
  int i    = idx & 15;
  int lane = (idx >> 4) & 31;
  int c    = (idx >> 9) & 3;
  int j    = idx >> 11;
  int n = j * 16 + (lane & 15);
  int k = c * 32 + ((lane >= 16) ? 8 : 0) + (i & 7) + ((i >= 8) ? 16 : 0);
  pk[idx] = (__bf16)W[n * 128 + k];
}

// ---------------------------------------------------------------------------
// Sparse scatter-add: out[b, dst[e], :] += feats[b, src[e], :]
// grid = (E*32/256, B); one float4 + 4 f32 atomics per thread.
// ---------------------------------------------------------------------------
__global__ void spmm_scatter(const float* __restrict__ feats,
                             const int* __restrict__ src, const int* __restrict__ dst,
                             float* __restrict__ out, int nsrc, int nout) {
  int idx = blockIdx.x * blockDim.x + threadIdx.x;
  int dg = idx & 31;          // 4-float group within D=128
  int e  = idx >> 5;
  int b  = blockIdx.y;
  int s = src[e];
  int d = dst[e];
  const float4 v = *(const float4*)(feats + ((size_t)b * nsrc + s) * D_ + dg * 4);
  float* op = out + ((size_t)b * nout + d) * D_ + dg * 4;
  atomicAdd(op + 0, v.x);
  atomicAdd(op + 1, v.y);
  atomicAdd(op + 2, v.z);
  atomicAdd(op + 3, v.w);
}

// ---------------------------------------------------------------------------
// Fused  msg_net (GEMM+GELU+GEMM)  ->  GRU cell, 16 rows per wave.
//   S  : spmm output, (R,128) f32
//   H  : previous hidden state, (R,128) f32
//   out: new hidden state, (R,128) f32
// Dynamic LDS: 2048 floats (8 KB) per wave, 8 waves/block -> 64 KB.
// ---------------------------------------------------------------------------
__global__ void __launch_bounds__(256)
fused_msg_gru(const float* __restrict__ S, const float* __restrict__ H,
              const bf16_t* __restrict__ pkW1, const bf16_t* __restrict__ pkW2,
              const float* __restrict__ b1, const float* __restrict__ b2,
              const bf16_t* __restrict__ pkIh, const bf16_t* __restrict__ pkHh,
              const float* __restrict__ bih, const float* __restrict__ bhh,
              float* __restrict__ out, int numTiles) {
  extern __shared__ float smem[];
  const int wave = threadIdx.x >> 5;
  const int lane = threadIdx.x & 31;
  const int tile = blockIdx.x * 8 + wave;
  if (tile >= numTiles) return;

  float* buf = smem + wave * 2048;        // 16 x 128 f32 staging, reused
  const int nl = lane & 15;
  const int o  = (lane >= 16) ? 8 : 0;    // A-operand K offset
  const int ms = (lane >= 16) ? 8 : 0;    // C-layout M offset
  const size_t rowBase = (size_t)tile * 16;

  // ---- A-operand: s rows (f32 -> bf16) ------------------------------------
  v16bf As[4];
  {
    const float* sr = S + (rowBase + nl) * D_;
#pragma unroll
    for (int c = 0; c < 4; ++c) As[c] = load_a16(sr + c * 32, o);
  }

  // ---- t = gelu(s @ w1^T + b1) -> LDS -------------------------------------
#pragma unroll 2
  for (int j = 0; j < 8; ++j) {
    v8f acc = gemm4(As, pkW1, j, lane);
    const float bb = b1[j * 16 + nl];
#pragma unroll
    for (int e = 0; e < 8; ++e)
      buf[(e + ms) * 128 + j * 16 + nl] = fast_gelu(acc[e] + bb);
  }
  __threadfence_block();

  // ---- A-operand: t from LDS ----------------------------------------------
  v16bf At[4];
  {
    const float* tr = buf + nl * 128;
#pragma unroll
    for (int c = 0; c < 4; ++c) At[c] = load_a16(tr + c * 32, o);
  }

  // ---- m = t @ w2^T + b2 -> LDS (DS in-order: WAR on buf is safe) ---------
#pragma unroll 2
  for (int j = 0; j < 8; ++j) {
    v8f acc = gemm4(At, pkW2, j, lane);
    const float bb = b2[j * 16 + nl];
#pragma unroll
    for (int e = 0; e < 8; ++e)
      buf[(e + ms) * 128 + j * 16 + nl] = acc[e] + bb;
  }
  __threadfence_block();

  // ---- A-operands: m (LDS) and h (global) ---------------------------------
  v16bf Am[4], Ah[4];
  {
    const float* mr = buf + nl * 128;
    const float* hr = H + (rowBase + nl) * D_;
#pragma unroll
    for (int c = 0; c < 4; ++c) {
      Am[c] = load_a16(mr + c * 32, o);
      Ah[c] = load_a16(hr + c * 32, o);
    }
  }

  // ---- GRU gates, 16 output columns at a time -----------------------------
  for (int j = 0; j < 8; ++j) {
    v8f giR = gemm4(Am, pkIh, j,      lane);
    v8f giZ = gemm4(Am, pkIh, j + 8,  lane);
    v8f giN = gemm4(Am, pkIh, j + 16, lane);
    v8f ghR = gemm4(Ah, pkHh, j,      lane);
    v8f ghZ = gemm4(Ah, pkHh, j + 8,  lane);
    v8f ghN = gemm4(Ah, pkHh, j + 16, lane);

    const float biR = bih[j * 16 + nl];
    const float biZ = bih[128 + j * 16 + nl];
    const float biN = bih[256 + j * 16 + nl];
    const float bhR = bhh[j * 16 + nl];
    const float bhZ = bhh[128 + j * 16 + nl];
    const float bhN = bhh[256 + j * 16 + nl];

#pragma unroll
    for (int e = 0; e < 8; ++e) {
      const float r = fast_sigmoid(giR[e] + biR + ghR[e] + bhR);
      const float z = fast_sigmoid(giZ[e] + biZ + ghZ[e] + bhZ);
      const float n = fast_tanh(giN[e] + biN + r * (ghN[e] + bhN));
      const size_t off = (rowBase + e + ms) * D_ + j * 16 + nl;
      const float hv = H[off];
      out[off] = (1.0f - z) * n + z * hv;
    }
  }
}

// ---------------------------------------------------------------------------
extern "C" void kernel_launch(void* const* d_in, const int* in_sizes, int n_in,
                              void* d_out, int out_size, void* d_ws, size_t ws_size,
                              hipStream_t stream) {
  (void)in_sizes; (void)n_in; (void)out_size; (void)ws_size;

  const float* v_feats  = (const float*)d_in[0];
  const float* c_feats  = (const float*)d_in[1];
  const int*   edge_chk = (const int*)d_in[2];
  const int*   edge_var = (const int*)d_in[3];
  const float* msg_w1   = (const float*)d_in[4];
  const float* msg_b1   = (const float*)d_in[5];
  const float* msg_w2   = (const float*)d_in[6];
  const float* msg_b2   = (const float*)d_in[7];
  const float* var_w_ih = (const float*)d_in[8];
  const float* var_w_hh = (const float*)d_in[9];
  const float* var_b_ih = (const float*)d_in[10];
  const float* var_b_hh = (const float*)d_in[11];
  const float* chk_w_ih = (const float*)d_in[12];
  const float* chk_w_hh = (const float*)d_in[13];
  const float* chk_b_ih = (const float*)d_in[14];
  const float* chk_b_hh = (const float*)d_in[15];

  float* v_new = (float*)d_out;                          // (B, NV, D)
  float* c_new = v_new + (size_t)B_ * NV_ * D_;          // (B, NC, D)

  // Workspace: [ Sbuf (B*NV*D f32, reused for both directions) | packed bf16 weights ]
  char*  ws   = (char*)d_ws;
  float* Sbuf = (float*)ws;
  bf16_t* pk     = (bf16_t*)(ws + (size_t)B_ * NV_ * D_ * sizeof(float));
  bf16_t* pkW1   = pk;
  bf16_t* pkW2   = pkW1 + 128 * 128;
  bf16_t* pkChkI = pkW2 + 128 * 128;
  bf16_t* pkChkH = pkChkI + 384 * 128;
  bf16_t* pkVarI = pkChkH + 384 * 128;
  bf16_t* pkVarH = pkVarI + 384 * 128;

  // --- pack weights (deterministic, every call) ---
  pack_w<<<64, 256, 0, stream>>>(msg_w1,   pkW1,   128 * 128);
  pack_w<<<64, 256, 0, stream>>>(msg_w2,   pkW2,   128 * 128);
  pack_w<<<192, 256, 0, stream>>>(chk_w_ih, pkChkI, 384 * 128);
  pack_w<<<192, 256, 0, stream>>>(chk_w_hh, pkChkH, 384 * 128);
  pack_w<<<192, 256, 0, stream>>>(var_w_ih, pkVarI, 384 * 128);
  pack_w<<<192, 256, 0, stream>>>(var_w_hh, pkVarH, 384 * 128);

  const dim3 gridS(E_ * 32 / 256, B_);
  const size_t shmem = 8 * 2048 * sizeof(float);   // 64 KB / block

  // --- direction 1: vars -> checks ---
  hipMemsetAsync(Sbuf, 0, (size_t)B_ * NC_ * D_ * sizeof(float), stream);
  spmm_scatter<<<gridS, 256, 0, stream>>>(v_feats, edge_var, edge_chk, Sbuf, NV_, NC_);
  fused_msg_gru<<<(B_ * NC_ / 16) / 8, 256, shmem, stream>>>(
      Sbuf, c_feats, pkW1, pkW2, msg_b1, msg_b2,
      pkChkI, pkChkH, chk_b_ih, chk_b_hh, c_new, B_ * NC_ / 16);

  // --- direction 2: checks -> vars ---
  hipMemsetAsync(Sbuf, 0, (size_t)B_ * NV_ * D_ * sizeof(float), stream);
  spmm_scatter<<<gridS, 256, 0, stream>>>(c_new, edge_chk, edge_var, Sbuf, NC_, NV_);
  fused_msg_gru<<<(B_ * NV_ / 16) / 8, 256, shmem, stream>>>(
      Sbuf, v_feats, pkW1, pkW2, msg_b1, msg_b2,
      pkVarI, pkVarH, var_b_ih, var_b_hh, v_new, B_ * NV_ / 16);
}
